// ETNN_30305289240604
// MI455X (gfx1250) — matrix operand order repaired and checked by reading, not compiled, not run on hardware
//
#include <hip/hip_runtime.h>
#include <hip/hip_bf16.h>
#include <cstdint>
#include <cstddef>

// ============================================================================
// ETNN forward for MI455X (gfx1250, wave32, WMMA).
// All dense layers run as bf16 WMMA (v_wmma_f32_16x16x32_bf16) with f32 accum.
// Weights are re-packed each launch into a contiguous-k (16*(lane>>4) + j) 32x16 B-fragment layout so
// the GEMM hot loop is ds_load_b128 (A) + global_load_b128 (B) + v_wmma.
// ============================================================================

typedef __bf16 bf16;
typedef __attribute__((ext_vector_type(16))) __bf16 v16bf;
typedef __attribute__((ext_vector_type(8)))  float  v8f;

#define N0C 50000
#define N1C 100000
#define N2C 50000
#define EC  100000
#define BC  64

// LDS row strides (bf16 elements), padded for bank spread + 16B alignment
#define SR_A 296   // message input rows: 256 (x_s|x_r) + 3 inv + pad -> 9 K-chunks
#define SR_M 136   // H=128 rows
#define SR_U 264   // update input rows: 256 -> 8 K-chunks
#define SR_P 392   // post input rows: 384 -> 12 K-chunks

union BF16x16 { v16bf v; uint4 u[2]; };

__device__ __forceinline__ float siluf(float x) { return x / (1.f + __expf(-x)); }

// A fragment (16x32 bf16, M=lane&15): elems 0..15 = K[kc+koff .. +15],
// koff = (lane>=16) ? 16 : 0.
__device__ __forceinline__ v16bf load_a(const bf16* lane_row, int kc, int lane) {
  const bf16* p = lane_row + kc + ((lane >> 4) << 4);
  BF16x16 t;
  t.u[0] = *(const uint4*)(p);
  t.u[1] = *(const uint4*)(p + 8);
  return t.v;
}

// B fragment (32x16 bf16) pre-packed: 512 elems/fragment, 16 contiguous per lane.
__device__ __forceinline__ v16bf load_b(const bf16* frag, int lane) {
  const bf16* p = frag + (lane << 4);
  BF16x16 t;
  t.u[0] = *(const uint4*)(p);
  t.u[1] = *(const uint4*)(p + 8);
  return t.v;
}

// C/D layout: acc[t][v] is row M = v + 8*(lane>>4), col n = 16*t + (lane&15).
__device__ __forceinline__ void init_bias(v8f acc[8], const float* b, int n0) {
#pragma unroll
  for (int t = 0; t < 8; ++t) {
    float x = b[t * 16 + n0];
#pragma unroll
    for (int v = 0; v < 8; ++v) acc[t][v] = x;
  }
}

// 16 rows x 128 cols GEMM accumulate: A from (LDS) lane_row, B packed global.
__device__ __forceinline__ void mm_tiles(const bf16* lane_row, int kchunks,
                                         const bf16* Wp, v8f acc[8], int lane) {
  for (int kc = 0; kc < kchunks; ++kc) {
    v16bf a = load_a(lane_row, kc * 32, lane);
    const bf16* wb = Wp + ((size_t)kc << 12);  // kc * 8 fragments * 512
#pragma unroll
    for (int t = 0; t < 8; ++t) {
      v16bf b = load_b(wb + ((size_t)t << 9), lane);
      acc[t] = __builtin_amdgcn_wmma_f32_16x16x32_bf16(
          false, a, false, b, (short)0, acc[t], false, false);
    }
  }
}

__device__ __forceinline__ void silu_store_lds(const v8f acc[8], bf16* Mw, int lane) {
  int rsel = (lane >> 4) << 3;
  int n0 = lane & 15;
#pragma unroll
  for (int t = 0; t < 8; ++t)
#pragma unroll
    for (int v = 0; v < 8; ++v)
      Mw[(size_t)(v + rsel) * SR_M + t * 16 + n0] = (bf16)siluf(acc[t][v]);
}

__device__ __forceinline__ void cvt_store4(bf16* dst, float4 v) {
  union { bf16 b[4]; uint2 u; } p;
  p.b[0] = (bf16)v.x; p.b[1] = (bf16)v.y; p.b[2] = (bf16)v.z; p.b[3] = (bf16)v.w;
  *(uint2*)dst = p.u;
}

// ---------------------------------------------------------------------------
// Weight packing: f32 [Ksrc x 128] row-major -> bf16 B-fragment stream,
// K padded to Kdst (multiple of 32) with zeros.
// fragment f = kc*8 + t; within: lane holds K = kc*32 + 16*(lane>>4) + j,
// n = 16*t + (lane&15), 16 contiguous elements per lane.
// ---------------------------------------------------------------------------
__global__ void pack_w(const float* __restrict__ W, bf16* __restrict__ out,
                       int Ksrc, int N, int Kdst) {
  int o = blockIdx.x * 256 + threadIdx.x;
  int total = Kdst * N;
  if (o >= total) return;
  int frag = o >> 9, r = o & 511, lane = r >> 4, j = r & 15;
  int tcnt = N >> 4;
  int kc = frag / tcnt, t = frag - kc * tcnt;
  int k = kc * 32 + ((lane >> 4) << 4) + j;
  int n = t * 16 + (lane & 15);
  float v = (k < Ksrc) ? W[(size_t)k * N + n] : 0.f;
  out[o] = (bf16)v;
}

// ---------------------------------------------------------------------------
// Geometry
// ---------------------------------------------------------------------------
__global__ void geom1_kernel(const float* __restrict__ pos, const int* __restrict__ cell,
                             float* __restrict__ cent, float* __restrict__ diam, int n) {
  int i = blockIdx.x * 256 + threadIdx.x;
  if (i >= n) return;
  int a = cell[2 * i], b = cell[2 * i + 1];
  float ax = pos[3 * a], ay = pos[3 * a + 1], az = pos[3 * a + 2];
  float bx = pos[3 * b], by = pos[3 * b + 1], bz = pos[3 * b + 2];
  cent[3 * i] = 0.5f * (ax + bx); cent[3 * i + 1] = 0.5f * (ay + by); cent[3 * i + 2] = 0.5f * (az + bz);
  float dx = ax - bx, dy = ay - by, dz = az - bz;
  diam[i] = sqrtf(dx * dx + dy * dy + dz * dz);
}

__global__ void geom2_kernel(const float* __restrict__ pos, const int* __restrict__ cell,
                             float* __restrict__ cent, float* __restrict__ diam, int n) {
  int i = blockIdx.x * 256 + threadIdx.x;
  if (i >= n) return;
  int a = cell[3 * i], b = cell[3 * i + 1], c = cell[3 * i + 2];
  float ax = pos[3 * a], ay = pos[3 * a + 1], az = pos[3 * a + 2];
  float bx = pos[3 * b], by = pos[3 * b + 1], bz = pos[3 * b + 2];
  float cx = pos[3 * c], cy = pos[3 * c + 1], cz = pos[3 * c + 2];
  const float third = 1.f / 3.f;
  cent[3 * i] = (ax + bx + cx) * third;
  cent[3 * i + 1] = (ay + by + cy) * third;
  cent[3 * i + 2] = (az + bz + cz) * third;
  float d0 = (ax-bx)*(ax-bx)+(ay-by)*(ay-by)+(az-bz)*(az-bz);
  float d1 = (ax-cx)*(ax-cx)+(ay-cy)*(ay-cy)+(az-cz)*(az-cz);
  float d2 = (bx-cx)*(bx-cx)+(by-cy)*(by-cy)+(bz-cz)*(bz-cz);
  diam[i] = sqrtf(fmaxf(d0, fmaxf(d1, d2)));
}

// ---------------------------------------------------------------------------
// Invariants + BN statistics (wave reduce -> atomics into stats[6])
// ---------------------------------------------------------------------------
__global__ void inv_kernel(const float* __restrict__ cs, const float* __restrict__ ds,
                           const float* __restrict__ cr, const float* __restrict__ dr,
                           const int* __restrict__ snd, const int* __restrict__ rcv,
                           float* __restrict__ feat, float* __restrict__ stats, int E) {
  int i = blockIdx.x * 256 + threadIdx.x;
  float f0 = 0.f, f1 = 0.f, f2 = 0.f;
  if (i < E) {
    int s = snd[i], r = rcv[i];
    float dx = cs[3 * s] - cr[3 * r];
    float dy = cs[3 * s + 1] - cr[3 * r + 1];
    float dz = cs[3 * s + 2] - cr[3 * r + 2];
    f0 = sqrtf(dx * dx + dy * dy + dz * dz);
    f1 = ds ? ds[s] : 0.f;
    f2 = dr ? dr[r] : 0.f;
    feat[(size_t)i * 3 + 0] = f0;
    feat[(size_t)i * 3 + 1] = f1;
    feat[(size_t)i * 3 + 2] = f2;
  }
  float v[6] = {f0, f1, f2, f0 * f0, f1 * f1, f2 * f2};
  for (int m = 16; m; m >>= 1)
#pragma unroll
    for (int k = 0; k < 6; ++k) v[k] += __shfl_xor(v[k], m, 32);
  if ((threadIdx.x & 31) == 0)
#pragma unroll
    for (int k = 0; k < 6; ++k) atomicAdd(&stats[k], v[k]);
}

struct BN5 { const float* g[5]; const float* b[5]; };

__global__ void bn_finalize(const float* __restrict__ stats, BN5 bn,
                            float* __restrict__ affine, float invE) {
  int i = threadIdx.x;
  if (i >= 15) return;
  int a = i / 3, f = i - a * 3;
  float mean = stats[a * 6 + f] * invE;
  float var = stats[a * 6 + 3 + f] * invE - mean * mean;
  float sc = bn.g[a][f] * rsqrtf(var + 1e-5f);
  affine[i * 2 + 0] = sc;
  affine[i * 2 + 1] = bn.b[a][f] - mean * sc;
}

// ---------------------------------------------------------------------------
// Embedding: x[row] @ W(16x128) + b   (tiny K -> plain VALU)
// ---------------------------------------------------------------------------
__global__ void embed_kernel(const float* __restrict__ xin, const float* __restrict__ W,
                             const float* __restrict__ b, float* __restrict__ xout, int N) {
  int i = blockIdx.x * 256 + threadIdx.x;
  if (i >= N * 128) return;
  int row = i >> 7, n = i & 127;
  const float* xr = xin + (size_t)row * 16;
  float s = b[n];
#pragma unroll
  for (int k = 0; k < 16; ++k) s += xr[k] * W[k * 128 + n];
  xout[i] = s;
}

// ---------------------------------------------------------------------------
// Fused message kernel: gather -> l1(259->128) silu -> l2(128->128) silu ->
// gate le(128->1) sigmoid -> scatter-add agg[rcv].  64 edges / block, 4 waves.
// ---------------------------------------------------------------------------
__launch_bounds__(128)
__global__ void msg_kernel(const float* __restrict__ xs, const float* __restrict__ xr,
                           const int* __restrict__ snd, const int* __restrict__ rcv,
                           const float* __restrict__ feat, const float* __restrict__ affine,
                           const bf16* __restrict__ W1, const float* __restrict__ b1,
                           const bf16* __restrict__ W2, const float* __restrict__ b2,
                           const float* __restrict__ wle, const float* __restrict__ ble,
                           float* __restrict__ agg, int E) {
  __shared__ bf16 A[64 * SR_A];
  __shared__ bf16 M[64 * SR_M];
  const int tid = threadIdx.x;
  const int lane = tid & 31;
  const int e0 = blockIdx.x * 64;
  {
    int r = tid >> 1, h = tid & 1;
    int e = e0 + r;
    int ec = (e < E) ? e : (E - 1);
    const float* src = (h == 0) ? (xs + (size_t)snd[ec] * 128) : (xr + (size_t)rcv[ec] * 128);
    bf16* dst = A + (size_t)r * SR_A + h * 128;
#pragma unroll 8
    for (int j = 0; j < 128; j += 4) cvt_store4(dst + j, *(const float4*)(src + j));
    if (h == 1) {
      bf16* d2 = A + (size_t)r * SR_A + 256;
      if (e < E) {
#pragma unroll
        for (int f = 0; f < 3; ++f)
          d2[f] = (bf16)(feat[(size_t)e * 3 + f] * affine[2 * f] + affine[2 * f + 1]);
      } else { d2[0] = (bf16)0.f; d2[1] = (bf16)0.f; d2[2] = (bf16)0.f; }
      for (int j = 3; j < 40; ++j) d2[j] = (bf16)0.f;   // zero K-pad up to 288
    }
  }
  __syncthreads();
  const int wv = tid >> 5;
  const int rowbase = wv * 16;
  const int n0 = lane & 15;
  v8f acc[8];
  init_bias(acc, b1, n0);
  mm_tiles(A + (size_t)(rowbase + n0) * SR_A, 9, W1, acc, lane);
  silu_store_lds(acc, M + (size_t)rowbase * SR_M, lane);
  v8f acc2[8];
  init_bias(acc2, b2, n0);
  mm_tiles(M + (size_t)(rowbase + n0) * SR_M, 4, W2, acc2, lane);
  // silu + gate dot (reduce over feature dim == lanes 0..15 of each half)
  float wl[8];
#pragma unroll
  for (int t = 0; t < 8; ++t) wl[t] = wle[t * 16 + n0];
  float p[8];
#pragma unroll
  for (int v = 0; v < 8; ++v) p[v] = 0.f;
#pragma unroll
  for (int t = 0; t < 8; ++t)
#pragma unroll
    for (int v = 0; v < 8; ++v) {
      float s = siluf(acc2[t][v]);
      acc2[t][v] = s;
      p[v] += s * wl[t];
    }
#pragma unroll
  for (int v = 0; v < 8; ++v)
    for (int m = 1; m < 16; m <<= 1) p[v] += __shfl_xor(p[v], m, 32);
  float lb = ble[0];
  int rsel = (lane >> 4) << 3;
#pragma unroll
  for (int v = 0; v < 8; ++v) {
    int row = rowbase + v + rsel;
    int e = e0 + row;
    if (e < E) {
      float g = 1.f / (1.f + __expf(-(p[v] + lb)));
      float* dst = agg + (size_t)rcv[e] * 128;
#pragma unroll
      for (int t = 0; t < 8; ++t) atomicAdd(dst + t * 16 + n0, acc2[t][v] * g);
    }
  }
}

// ---------------------------------------------------------------------------
// Update: x += l2( silu( l1([x|agg]) ) ), in place. 64 rows / block.
// ---------------------------------------------------------------------------
__launch_bounds__(128)
__global__ void upd_kernel(float* __restrict__ x, const float* __restrict__ agg,
                           const bf16* __restrict__ W1, const float* __restrict__ b1,
                           const bf16* __restrict__ W2, const float* __restrict__ b2, int N) {
  __shared__ bf16 A[64 * SR_U];
  __shared__ bf16 M[64 * SR_M];
  int tid = threadIdx.x, lane = tid & 31;
  int r0 = blockIdx.x * 64;
  {
    int r = tid >> 1, h = tid & 1;
    int row = r0 + r; if (row >= N) row = N - 1;
    const float* src = (h == 0) ? (x + (size_t)row * 128) : (agg + (size_t)row * 128);
    bf16* dst = A + (size_t)r * SR_U + h * 128;
#pragma unroll 8
    for (int j = 0; j < 128; j += 4) cvt_store4(dst + j, *(const float4*)(src + j));
  }
  __syncthreads();
  int wv = tid >> 5, rowbase = wv * 16, n0 = lane & 15;
  v8f acc[8];
  init_bias(acc, b1, n0);
  mm_tiles(A + (size_t)(rowbase + n0) * SR_U, 8, W1, acc, lane);
  silu_store_lds(acc, M + (size_t)rowbase * SR_M, lane);
  v8f acc2[8];
  init_bias(acc2, b2, n0);
  mm_tiles(M + (size_t)(rowbase + n0) * SR_M, 4, W2, acc2, lane);
  int rsel = (lane >> 4) << 3;
#pragma unroll
  for (int v = 0; v < 8; ++v) {
    int row = r0 + rowbase + v + rsel;
    if (row < N) {
      float* xr = x + (size_t)row * 128;
#pragma unroll
      for (int t = 0; t < 8; ++t) { int o = t * 16 + n0; xr[o] = xr[o] + acc2[t][v]; }
    }
  }
}

// ---------------------------------------------------------------------------
// Pre-pool: f = l2(silu(l1(x))); pooled[batch[row], rankoff + n] += f
// ---------------------------------------------------------------------------
__launch_bounds__(128)
__global__ void prepool_kernel(const float* __restrict__ x, const int* __restrict__ batch,
                               const bf16* __restrict__ W1, const float* __restrict__ b1,
                               const bf16* __restrict__ W2, const float* __restrict__ b2,
                               float* __restrict__ pooled, int rankoff, int N) {
  __shared__ bf16 A[64 * SR_M];
  __shared__ bf16 M[64 * SR_M];
  int tid = threadIdx.x, lane = tid & 31;
  int r0 = blockIdx.x * 64;
  {
    int r = tid >> 1, h = tid & 1;
    int row = r0 + r; if (row >= N) row = N - 1;
    const float* src = x + (size_t)row * 128 + h * 64;
    bf16* dst = A + (size_t)r * SR_M + h * 64;
#pragma unroll 4
    for (int j = 0; j < 64; j += 4) cvt_store4(dst + j, *(const float4*)(src + j));
  }
  __syncthreads();
  int wv = tid >> 5, rowbase = wv * 16, n0 = lane & 15;
  v8f acc[8];
  init_bias(acc, b1, n0);
  mm_tiles(A + (size_t)(rowbase + n0) * SR_M, 4, W1, acc, lane);
  silu_store_lds(acc, M + (size_t)rowbase * SR_M, lane);
  v8f acc2[8];
  init_bias(acc2, b2, n0);
  mm_tiles(M + (size_t)(rowbase + n0) * SR_M, 4, W2, acc2, lane);
  int rsel = (lane >> 4) << 3;
#pragma unroll
  for (int v = 0; v < 8; ++v) {
    int row = r0 + rowbase + v + rsel;
    if (row < N) {
      float* dst = pooled + (size_t)batch[row] * 384 + rankoff;
#pragma unroll
      for (int t = 0; t < 8; ++t) atomicAdd(dst + t * 16 + n0, acc2[t][v]);
    }
  }
}

// ---------------------------------------------------------------------------
// Post MLP (B=64 rows): split into two kernels to keep LDS < 64KB each.
// ---------------------------------------------------------------------------
__launch_bounds__(128)
__global__ void post1_kernel(const float* __restrict__ pooled, const bf16* __restrict__ W1,
                             const float* __restrict__ b1, float* __restrict__ hid) {
  __shared__ bf16 A[64 * SR_P];
  int tid = threadIdx.x, lane = tid & 31;
  {
    int r = tid >> 1, h = tid & 1;
    const float* src = pooled + (size_t)r * 384 + h * 192;
    bf16* dst = A + (size_t)r * SR_P + h * 192;
#pragma unroll 8
    for (int j = 0; j < 192; j += 4) cvt_store4(dst + j, *(const float4*)(src + j));
  }
  __syncthreads();
  int wv = tid >> 5, rowbase = wv * 16, n0 = lane & 15;
  v8f acc[8];
  init_bias(acc, b1, n0);
  mm_tiles(A + (size_t)(rowbase + n0) * SR_P, 12, W1, acc, lane);
  int rsel = (lane >> 4) << 3;
#pragma unroll
  for (int t = 0; t < 8; ++t)
#pragma unroll
    for (int v = 0; v < 8; ++v)
      hid[(size_t)(rowbase + v + rsel) * 128 + t * 16 + n0] = siluf(acc[t][v]);
}

__launch_bounds__(128)
__global__ void post2_kernel(const float* __restrict__ hid, const bf16* __restrict__ W2,
                             const float* __restrict__ b2, float* __restrict__ out) {
  __shared__ bf16 A[64 * SR_M];
  int tid = threadIdx.x, lane = tid & 31;
  {
    int r = tid >> 1, h = tid & 1;
    const float* src = hid + (size_t)r * 128 + h * 64;
    bf16* dst = A + (size_t)r * SR_M + h * 64;
#pragma unroll 4
    for (int j = 0; j < 64; j += 4) cvt_store4(dst + j, *(const float4*)(src + j));
  }
  __syncthreads();
  int wv = tid >> 5, rowbase = wv * 16, n0 = lane & 15;
  v8f acc[8];
  init_bias(acc, b2, n0);
  mm_tiles(A + (size_t)(rowbase + n0) * SR_M, 4, W2, acc, lane);
  int rsel = (lane >> 4) << 3;
#pragma unroll
  for (int t = 0; t < 8; ++t)
#pragma unroll
    for (int v = 0; v < 8; ++v)
      out[(size_t)(rowbase + v + rsel) * 128 + t * 16 + n0] = acc[t][v];
}

// ============================================================================
// Host launcher
// ============================================================================
extern "C" void kernel_launch(void* const* d_in, const int* in_sizes, int n_in,
                              void* d_out, int out_size, void* d_ws, size_t ws_size,
                              hipStream_t stream) {
  (void)in_sizes; (void)n_in; (void)out_size; (void)ws_size;
  const int Nsz[3] = {N0C, N1C, N2C};

  // ---- inputs (setup_inputs order) ----
  const float* pos = (const float*)d_in[0];
  const float* xin[3] = {(const float*)d_in[1], (const float*)d_in[2], (const float*)d_in[3]};
  const int* cell1 = (const int*)d_in[4];
  const int* cell2 = (const int*)d_in[5];
  const int* snd[5]; const int* rcv[5];
  for (int a = 0; a < 5; ++a) { snd[a] = (const int*)d_in[6 + 2 * a]; rcv[a] = (const int*)d_in[7 + 2 * a]; }
  const int* batch[3] = {(const int*)d_in[16], (const int*)d_in[17], (const int*)d_in[18]};

  // ---- params: JAX pytree flatten order (dict keys sorted at every level) ----
  // params{'bn','emb','layers','post','pre'} -> bn, emb, layers, post, pre
  // lin {'w','b'} -> b, w ; msg {'l1','l2','le'} -> l1, l2, le
  const float *bn_b[5], *bn_g[5];
  for (int a = 0; a < 5; ++a) { bn_b[a] = (const float*)d_in[19 + 2 * a]; bn_g[a] = (const float*)d_in[20 + 2 * a]; }
  const float *emb_b[3], *emb_w[3];
  for (int r = 0; r < 3; ++r) { emb_b[r] = (const float*)d_in[29 + 2 * r]; emb_w[r] = (const float*)d_in[30 + 2 * r]; }
  const float *m_b1[2][5], *m_w1[2][5], *m_b2[2][5], *m_w2[2][5], *m_ble[2][5], *m_wle[2][5];
  const float *u_b1[2][3], *u_w1[2][3], *u_b2[2][3], *u_w2[2][3];
  for (int l = 0; l < 2; ++l) {
    for (int a = 0; a < 5; ++a) {
      int base = 35 + l * 42 + a * 6;
      m_b1[l][a] = (const float*)d_in[base + 0]; m_w1[l][a] = (const float*)d_in[base + 1];
      m_b2[l][a] = (const float*)d_in[base + 2]; m_w2[l][a] = (const float*)d_in[base + 3];
      m_ble[l][a] = (const float*)d_in[base + 4]; m_wle[l][a] = (const float*)d_in[base + 5];
    }
    for (int r = 0; r < 3; ++r) {
      int base = 35 + l * 42 + 30 + r * 4;
      u_b1[l][r] = (const float*)d_in[base + 0]; u_w1[l][r] = (const float*)d_in[base + 1];
      u_b2[l][r] = (const float*)d_in[base + 2]; u_w2[l][r] = (const float*)d_in[base + 3];
    }
  }
  const float* post_b1 = (const float*)d_in[119]; const float* post_w1 = (const float*)d_in[120];
  const float* post_b2 = (const float*)d_in[121]; const float* post_w2 = (const float*)d_in[122];
  const float *p_b1[3], *p_w1[3], *p_b2[3], *p_w2[3];
  for (int r = 0; r < 3; ++r) {
    int base = 123 + r * 4;
    p_b1[r] = (const float*)d_in[base + 0]; p_w1[r] = (const float*)d_in[base + 1];
    p_b2[r] = (const float*)d_in[base + 2]; p_w2[r] = (const float*)d_in[base + 3];
  }

  // ---- workspace carve-up ----
  char* wsb = (char*)d_ws;
  size_t cur = 0;
  auto carve = [&](size_t bytes) -> void* {
    void* p = wsb + cur;
    cur = (cur + bytes + 255) & ~(size_t)255;
    return p;
  };
  float* xb[3]; float* ag[3];
  for (int r = 0; r < 3; ++r) xb[r] = (float*)carve((size_t)Nsz[r] * 128 * 4);
  for (int r = 0; r < 3; ++r) ag[r] = (float*)carve((size_t)Nsz[r] * 128 * 4);
  float* cent1 = (float*)carve((size_t)N1C * 3 * 4);
  float* cent2 = (float*)carve((size_t)N2C * 3 * 4);
  float* diam1 = (float*)carve((size_t)N1C * 4);
  float* diam2 = (float*)carve((size_t)N2C * 4);
  float* featb[5];
  for (int a = 0; a < 5; ++a) featb[a] = (float*)carve((size_t)EC * 3 * 4);
  float* stats = (float*)carve(32 * 4);
  float* affine = (float*)carve(32 * 4);
  float* pooled = (float*)carve((size_t)BC * 384 * 4);
  float* hid = (float*)carve((size_t)BC * 128 * 4);
  bf16 *pk_m1[2][5], *pk_m2[2][5], *pk_u1[2][3], *pk_u2[2][3], *pk_p1[3], *pk_p2[3], *pk_q1, *pk_q2;
  for (int l = 0; l < 2; ++l)
    for (int a = 0; a < 5; ++a) {
      pk_m1[l][a] = (bf16*)carve(288 * 128 * 2);
      pk_m2[l][a] = (bf16*)carve(128 * 128 * 2);
    }
  for (int l = 0; l < 2; ++l)
    for (int r = 0; r < 3; ++r) {
      pk_u1[l][r] = (bf16*)carve(256 * 128 * 2);
      pk_u2[l][r] = (bf16*)carve(128 * 128 * 2);
    }
  for (int r = 0; r < 3; ++r) {
    pk_p1[r] = (bf16*)carve(128 * 128 * 2);
    pk_p2[r] = (bf16*)carve(128 * 128 * 2);
  }
  pk_q1 = (bf16*)carve(384 * 128 * 2);
  pk_q2 = (bf16*)carve(128 * 128 * 2);

  auto packW = [&](const float* W, bf16* dst, int Ksrc, int Kdst) {
    int total = Kdst * 128;
    pack_w<<<(total + 255) / 256, 256, 0, stream>>>(W, dst, Ksrc, 128, Kdst);
  };

  // ---- pack all GEMM weights into WMMA B-fragment layout ----
  for (int l = 0; l < 2; ++l) {
    for (int a = 0; a < 5; ++a) {
      packW(m_w1[l][a], pk_m1[l][a], 259, 288);
      packW(m_w2[l][a], pk_m2[l][a], 128, 128);
    }
    for (int r = 0; r < 3; ++r) {
      packW(u_w1[l][r], pk_u1[l][r], 256, 256);
      packW(u_w2[l][r], pk_u2[l][r], 128, 128);
    }
  }
  for (int r = 0; r < 3; ++r) {
    packW(p_w1[r], pk_p1[r], 128, 128);
    packW(p_w2[r], pk_p2[r], 128, 128);
  }
  packW(post_w1, pk_q1, 384, 384);
  packW(post_w2, pk_q2, 128, 128);

  // ---- geometry ----
  geom1_kernel<<<(N1C + 255) / 256, 256, 0, stream>>>(pos, cell1, cent1, diam1, N1C);
  geom2_kernel<<<(N2C + 255) / 256, 256, 0, stream>>>(pos, cell2, cent2, diam2, N2C);

  // ---- embeddings ----
  for (int r = 0; r < 3; ++r)
    embed_kernel<<<(Nsz[r] * 128 + 255) / 256, 256, 0, stream>>>(xin[r], emb_w[r], emb_b[r], xb[r], Nsz[r]);

  // ---- invariants + batchnorm stats ----
  hipMemsetAsync(stats, 0, 32 * 4, stream);
  const float* centp[3] = {pos, cent1, cent2};
  const float* diamp[3] = {nullptr, diam1, diam2};
  const int srk[5] = {0, 0, 1, 1, 2}, rrk[5] = {0, 1, 1, 2, 2};
  for (int a = 0; a < 5; ++a)
    inv_kernel<<<(EC + 255) / 256, 256, 0, stream>>>(centp[srk[a]], diamp[srk[a]],
                                                     centp[rrk[a]], diamp[rrk[a]],
                                                     snd[a], rcv[a], featb[a], stats + a * 6, EC);
  BN5 bn;
  for (int a = 0; a < 5; ++a) { bn.g[a] = bn_g[a]; bn.b[a] = bn_b[a]; }
  bn_finalize<<<1, 32, 0, stream>>>(stats, bn, affine, 1.0f / EC);

  // ---- message-passing layers ----
  const int EB = (EC + 63) / 64;
  for (int l = 0; l < 2; ++l) {
    for (int r = 0; r < 3; ++r)
      hipMemsetAsync(ag[r], 0, (size_t)Nsz[r] * 128 * 4, stream);
    for (int a = 0; a < 5; ++a)
      msg_kernel<<<EB, 128, 0, stream>>>(xb[srk[a]], xb[rrk[a]], snd[a], rcv[a],
                                         featb[a], affine + a * 6,
                                         pk_m1[l][a], m_b1[l][a], pk_m2[l][a], m_b2[l][a],
                                         m_wle[l][a], m_ble[l][a], ag[rrk[a]], EC);
    for (int r = 0; r < 3; ++r)
      upd_kernel<<<(Nsz[r] + 63) / 64, 128, 0, stream>>>(xb[r], ag[r],
                                                         pk_u1[l][r], u_b1[l][r],
                                                         pk_u2[l][r], u_b2[l][r], Nsz[r]);
  }

  // ---- pre-pool MLPs + global add-pool ----
  hipMemsetAsync(pooled, 0, (size_t)BC * 384 * 4, stream);
  for (int r = 0; r < 3; ++r)
    prepool_kernel<<<(Nsz[r] + 63) / 64, 128, 0, stream>>>(xb[r], batch[r],
                                                           pk_p1[r], p_b1[r],
                                                           pk_p2[r], p_b2[r],
                                                           pooled, r * 128, Nsz[r]);

  // ---- post MLP -> d_out [64,128] ----
  post1_kernel<<<1, 128, 0, stream>>>(pooled, pk_q1, post_b1, hid);
  post2_kernel<<<1, 128, 0, stream>>>(hid, pk_q2, post_b2, (float*)d_out);
}
